// HANLayer_17798344474649
// MI455X (gfx1250) — matrix operand order repaired and checked by reading, not compiled, not run on hardware
//
#include <hip/hip_runtime.h>
#include <hip/hip_bf16.h>
#include <math.h>

#define NNODES   50000
#define NEDGES   800000
#define IN_DIM_  128
#define OUT_DIM_ 64
#define HEADS_   4
#define HC_      256          // HEADS*OUT_DIM
#define NEG_SLOPE_ 0.2f

typedef __attribute__((ext_vector_type(16))) __bf16 v16bf;
typedef __attribute__((ext_vector_type(8)))  float  v8f;

union ABFrag { v16bf v; uint4 q[2]; };

__device__ __forceinline__ unsigned short f2bf(float f) {
  unsigned int u = __float_as_uint(f);
  unsigned int r = u + 0x7FFFu + ((u >> 16) & 1u);  // round-nearest-even
  return (unsigned short)(r >> 16);
}
__device__ __forceinline__ float bf2f(unsigned short s) {
  return __uint_as_float(((unsigned int)s) << 16);
}
__device__ __forceinline__ void atomicMaxF(float* addr, float v) {
  if (v >= 0.f) atomicMax((int*)addr, __float_as_int(v));
  else          atomicMin((unsigned int*)addr, (unsigned int)__float_as_int(v));
}

// ---------------- precision conversion ----------------
__global__ void k_x_to_bf(const float* __restrict__ x, unsigned short* __restrict__ xbf, int n) {
  int i = blockIdx.x * blockDim.x + threadIdx.x;
  if (i < n) xbf[i] = f2bf(x[i]);
}

// W [128,256] fp32 row-major -> Wt [256,128] bf16 (column-major for contiguous B frags)
__global__ void k_w_to_bf_t(const float* __restrict__ W, unsigned short* __restrict__ wt) {
  int i = blockIdx.x * blockDim.x + threadIdx.x;   // 32768 exact
  int k = i >> 8, c = i & 255;
  wt[c * IN_DIM_ + k] = f2bf(W[k * HC_ + c]);
}

__global__ void k_init(float* __restrict__ emax, float* __restrict__ denom, float* __restrict__ z) {
  int i = blockIdx.x * blockDim.x + threadIdx.x;   // N*64 exact
  if (i < NNODES * HEADS_) { emax[i] = -INFINITY; denom[i] = 0.f; }
  z[i] = 0.f;
}

// ---------------- h = x @ W  (bf16 WMMA, f32 accum), stored bf16 ----------------
// grid = N/16 blocks, 128 threads (4 waves). Wave w computes cols [64w,64w+64) as
// four 16x16 tiles, K-loop 128 in 4 steps of 32.
__global__ void __launch_bounds__(128) k_gemm(const unsigned short* __restrict__ xbf,
                                              const unsigned short* __restrict__ wt,
                                              unsigned short* __restrict__ hbf) {
  const int lane = threadIdx.x & 31;
  const int wave = threadIdx.x >> 5;
  const int rm   = lane & 15;      // row (A) / col (B,D) within tile
  const int half = lane >> 4;      // K-half selector per ISA layouts
  const int row0 = blockIdx.x * 16;

  v8f acc[4] = {};
  const unsigned short* arow = xbf + (size_t)(row0 + rm) * IN_DIM_;

  #pragma unroll
  for (int ks = 0; ks < IN_DIM_; ks += 32) {
    // A 16x32 bf16 frag: elems i -> K=ks+half*16+i
    ABFrag a;
    a.q[0] = *(const uint4*)(arow + ks + half * 16);
    a.q[1] = *(const uint4*)(arow + ks + half * 16 + 8);
    #pragma unroll
    for (int t = 0; t < 4; ++t) {
      int col = wave * 64 + t * 16 + rm;
      // B 32x16 bf16 frag: lane holds column, elems i -> K = ks + half*16 + i
      const unsigned short* bcol = wt + (size_t)col * IN_DIM_ + ks + half * 16;
      ABFrag b;
      b.q[0] = ((const uint4*)bcol)[0];
      b.q[1] = ((const uint4*)bcol)[1];
      acc[t] = __builtin_amdgcn_wmma_f32_16x16x32_bf16(
          false, a.v, false, b.v, (short)0, acc[t], false, false);
    }
  }
  // D 16x16 f32: VGPR r -> row = r + 8*half, col = rm
  #pragma unroll
  for (int t = 0; t < 4; ++t) {
    int col = wave * 64 + t * 16 + rm;
    #pragma unroll
    for (int r = 0; r < 8; ++r) {
      int row = row0 + half * 8 + r;
      hbf[(size_t)row * HC_ + col] = f2bf(acc[t][r]);
    }
  }
}

// ---------------- a_src/a_dst per (node, head) ----------------
__global__ void k_attn(const unsigned short* __restrict__ hbf,
                       const float* __restrict__ att_src, const float* __restrict__ att_dst,
                       float* __restrict__ a_src, float* __restrict__ a_dst) {
  int i = blockIdx.x * blockDim.x + threadIdx.x;
  if (i >= NNODES * HEADS_) return;
  int n = i >> 2, hd = i & 3;
  const unsigned short* hp = hbf + (size_t)n * HC_ + hd * OUT_DIM_;
  const float* as = att_src + hd * OUT_DIM_;
  const float* ad = att_dst + hd * OUT_DIM_;
  float s = 0.f, d = 0.f;
  #pragma unroll 8
  for (int c = 0; c < OUT_DIM_; ++c) { float h = bf2f(hp[c]); s += h * as[c]; d += h * ad[c]; }
  a_src[i] = s; a_dst[i] = d;
}

// ---------------- edge softmax pass 1: max ----------------
__global__ void k_edge_max(const int* __restrict__ ei, const float* __restrict__ a_src,
                           const float* __restrict__ a_dst, float* __restrict__ emax) {
  int e = blockIdx.x * blockDim.x + threadIdx.x;
  if (e >= NEDGES + NNODES) return;
  int s = (e < NEDGES) ? ei[e]          : (e - NEDGES);
  int d = (e < NEDGES) ? ei[NEDGES + e] : (e - NEDGES);
  #pragma unroll
  for (int hd = 0; hd < HEADS_; ++hd) {
    float v = a_src[s * 4 + hd] + a_dst[d * 4 + hd];
    v = (v >= 0.f) ? v : NEG_SLOPE_ * v;
    atomicMaxF(&emax[d * 4 + hd], v);
  }
}

// ---------------- edge softmax pass 2: denom ----------------
__global__ void k_edge_sum(const int* __restrict__ ei, const float* __restrict__ a_src,
                           const float* __restrict__ a_dst, const float* __restrict__ emax,
                           float* __restrict__ denom) {
  int e = blockIdx.x * blockDim.x + threadIdx.x;
  if (e >= NEDGES + NNODES) return;
  int s = (e < NEDGES) ? ei[e]          : (e - NEDGES);
  int d = (e < NEDGES) ? ei[NEDGES + e] : (e - NEDGES);
  #pragma unroll
  for (int hd = 0; hd < HEADS_; ++hd) {
    float v = a_src[s * 4 + hd] + a_dst[d * 4 + hd];
    v = (v >= 0.f) ? v : NEG_SLOPE_ * v;
    atomicAdd(&denom[d * 4 + hd], __expf(v - emax[d * 4 + hd]));
  }
}

// ---------------- aggregate: one wave per edge, head-mean folded in ----------------
__global__ void __launch_bounds__(256) k_agg(const int* __restrict__ ei,
      const float* __restrict__ a_src, const float* __restrict__ a_dst,
      const float* __restrict__ emax, const float* __restrict__ denom,
      const unsigned short* __restrict__ hbf, float* __restrict__ z) {
  int wid  = (blockIdx.x * blockDim.x + threadIdx.x) >> 5;
  int lane = threadIdx.x & 31;
  if (wid >= NEDGES + NNODES) return;
  int s = (wid < NEDGES) ? ei[wid]          : (wid - NEDGES);
  int d = (wid < NEDGES) ? ei[NEDGES + wid] : (wid - NEDGES);
  float alpha[HEADS_];
  #pragma unroll
  for (int hd = 0; hd < HEADS_; ++hd) {
    float v = a_src[s * 4 + hd] + a_dst[d * 4 + hd];
    v = (v >= 0.f) ? v : NEG_SLOPE_ * v;
    alpha[hd] = 0.25f * __expf(v - emax[d * 4 + hd]) / denom[d * 4 + hd];  // 0.25 = head mean
  }
  const unsigned short* hp = hbf + (size_t)s * HC_;
  int c0 = lane * 2;
  float acc0 = 0.f, acc1 = 0.f;
  #pragma unroll
  for (int hd = 0; hd < HEADS_; ++hd) {
    unsigned int u = *(const unsigned int*)(hp + hd * OUT_DIM_ + c0);
    acc0 += alpha[hd] * bf2f((unsigned short)(u & 0xFFFFu));
    acc1 += alpha[hd] * bf2f((unsigned short)(u >> 16));
  }
  atomicAdd(&z[(size_t)d * OUT_DIM_ + c0], acc0);
  atomicAdd(&z[(size_t)d * OUT_DIM_ + c0 + 1], acc1);
}

// ---------------- z = elu(z + bias) in place ----------------
__global__ void k_elu_bias(float* __restrict__ z, const float* __restrict__ bias) {
  int i = blockIdx.x * blockDim.x + threadIdx.x;   // N*64 exact
  float v = z[i] + bias[i & 63];
  z[i] = (v > 0.f) ? v : (__expf(v) - 1.f);
}

// ---------------- semantic attention: one wave per node ----------------
__global__ void __launch_bounds__(256) k_semantic(const float* __restrict__ zbuf,
      const float* __restrict__ projW, const float* __restrict__ projb,
      const float* __restrict__ av, float* __restrict__ zfinal, float* __restrict__ beta_out) {
  int node = (blockIdx.x * blockDim.x + threadIdx.x) >> 5;
  int lane = threadIdx.x & 31;
  if (node >= NNODES) return;
  float score[3], zlo[3] = {0.f, 0.f, 0.f}, zhi[3] = {0.f, 0.f, 0.f};
  #pragma unroll
  for (int m = 0; m < 3; ++m) {
    const float* zn = zbuf + ((size_t)m * NNODES + node) * OUT_DIM_;
    float acc0 = 0.f, acc1 = 0.f;
    for (int cp = 0; cp < OUT_DIM_; ++cp) {
      float zv = zn[cp];
      if (cp == lane)      zlo[m] = zv;
      if (cp == lane + 32) zhi[m] = zv;
      acc0 += zv * projW[cp * OUT_DIM_ + lane];
      acc1 += zv * projW[cp * OUT_DIM_ + lane + 32];
    }
    float w0 = tanhf(acc0 + projb[lane]);
    float w1 = tanhf(acc1 + projb[lane + 32]);
    float p  = w0 * av[lane] + w1 * av[lane + 32];
    #pragma unroll
    for (int off = 16; off > 0; off >>= 1) p += __shfl_xor(p, off, 32);
    score[m] = p;  // full sum in every lane (butterfly)
  }
  float mx = fmaxf(score[0], fmaxf(score[1], score[2]));
  float e0 = __expf(score[0] - mx), e1 = __expf(score[1] - mx), e2 = __expf(score[2] - mx);
  float inv = 1.f / (e0 + e1 + e2);
  float b0 = e0 * inv, b1 = e1 * inv, b2 = e2 * inv;
  zfinal[(size_t)node * OUT_DIM_ + lane]      = zlo[0] * b0 + zlo[1] * b1 + zlo[2] * b2;
  zfinal[(size_t)node * OUT_DIM_ + lane + 32] = zhi[0] * b0 + zhi[1] * b1 + zhi[2] * b2;
  if (lane < 3) beta_out[(size_t)node * 3 + lane] = (lane == 0) ? b0 : ((lane == 1) ? b1 : b2);
}

extern "C" void kernel_launch(void* const* d_in, const int* in_sizes, int n_in,
                              void* d_out, int out_size, void* d_ws, size_t ws_size,
                              hipStream_t stream) {
  (void)in_sizes; (void)n_in; (void)out_size; (void)ws_size;
  const float* x     = (const float*)d_in[0];
  const float* projW = (const float*)d_in[16];
  const float* projb = (const float*)d_in[17];
  const float* av    = (const float*)d_in[18];

  char* ws = (char*)d_ws;
  size_t off = 0;
  auto alloc = [&](size_t bytes) -> char* {
    char* p = ws + off; off += (bytes + 255) & ~(size_t)255; return p;
  };
  unsigned short* xbf = (unsigned short*)alloc((size_t)NNODES * IN_DIM_ * 2);  // 12.8 MB
  unsigned short* wt  = (unsigned short*)alloc((size_t)HC_ * IN_DIM_ * 2);     // 64 KB
  unsigned short* hbf = (unsigned short*)alloc((size_t)NNODES * HC_ * 2);      // 25.6 MB
  float* a_src = (float*)alloc((size_t)NNODES * HEADS_ * 4);
  float* a_dst = (float*)alloc((size_t)NNODES * HEADS_ * 4);
  float* emax  = (float*)alloc((size_t)NNODES * HEADS_ * 4);
  float* denom = (float*)alloc((size_t)NNODES * HEADS_ * 4);
  float* zbuf  = (float*)alloc((size_t)3 * NNODES * OUT_DIM_ * 4);             // 38.4 MB

  float* zfinal = (float*)d_out;
  float* beta   = zfinal + (size_t)NNODES * OUT_DIM_;

  k_x_to_bf<<<(NNODES * IN_DIM_ + 255) / 256, 256, 0, stream>>>(x, xbf, NNODES * IN_DIM_);

  const int EN = NEDGES + NNODES;
  for (int m = 0; m < 3; ++m) {
    const int*   ei   = (const int*)  d_in[1 + m * 5];
    const float* W    = (const float*)d_in[2 + m * 5];
    const float* attS = (const float*)d_in[3 + m * 5];
    const float* attD = (const float*)d_in[4 + m * 5];
    const float* bias = (const float*)d_in[5 + m * 5];
    float* zm = zbuf + (size_t)m * NNODES * OUT_DIM_;

    k_w_to_bf_t<<<(IN_DIM_ * HC_) / 256, 256, 0, stream>>>(W, wt);
    k_init<<<(NNODES * OUT_DIM_) / 256, 256, 0, stream>>>(emax, denom, zm);
    k_gemm<<<NNODES / 16, 128, 0, stream>>>(xbf, wt, hbf);
    k_attn<<<(NNODES * HEADS_ + 255) / 256, 256, 0, stream>>>(hbf, attS, attD, a_src, a_dst);
    k_edge_max<<<(EN + 255) / 256, 256, 0, stream>>>(ei, a_src, a_dst, emax);
    k_edge_sum<<<(EN + 255) / 256, 256, 0, stream>>>(ei, a_src, a_dst, emax, denom);
    k_agg<<<(EN * 32) / 256, 256, 0, stream>>>(ei, a_src, a_dst, emax, denom, hbf, zm);
    k_elu_bias<<<(NNODES * OUT_DIM_) / 256, 256, 0, stream>>>(zm, bias);
  }
  k_semantic<<<(NNODES * 32) / 256, 256, 0, stream>>>(zbuf, projW, projb, av, zfinal, beta);
}
